// CosinAttention_1211180778097
// MI455X (gfx1250) — matrix operand order repaired and checked by reading, not compiled, not run on hardware
//
#include <hip/hip_runtime.h>
#include <hip/hip_bf16.h>
#include <math.h>

typedef __attribute__((ext_vector_type(16))) _Float16 v16h;
typedef __attribute__((ext_vector_type(8)))  _Float16 v8h;
typedef __attribute__((ext_vector_type(8)))  float    v8f;

#define H_   16
#define S_   4096
#define D_   64
#define KB_  32           // keys per block-iteration
#define QT_  2            // query sub-tiles (16 rows each) per wave
#define PSTRH 40          // LDS stride (halfs) for P tile; multiple of 8 -> 16B-aligned chunks

// log(sqrt(64)) — clamp ceiling for logit_scale
#define MAX_LOG_SCALE 2.0794415416798357f
#define LOG2E 1.4426950408889634f

// ---------------------------------------------------------------------------
// Raw v_exp_f32 (exp2).  Our exponents are bounded in [-23.1, 0], far from the
// denormal threshold (-126), so libm exp2f's range-fixup (cmp+cndmask+ldexp)
// is dead weight — use the bare hardware instruction when available.
// ---------------------------------------------------------------------------
#if defined(__has_builtin)
#if __has_builtin(__builtin_amdgcn_exp2f)
#define COSATTN_EXP2(x) __builtin_amdgcn_exp2f(x)
#endif
#if __has_builtin(__builtin_amdgcn_update_dpp)
#define COSATTN_HAVE_DPP 1
#endif
#endif
#ifndef COSATTN_EXP2
#define COSATTN_EXP2(x) exp2f(x)
#endif

// ---------------------------------------------------------------------------
// 16-lane butterfly sum (lanes 0-15 and 16-31 reduce independently).
// DPP path: after xor1+xor2 each quad is uniform, so row_half_mirror (lane^7)
// acts as xor4; after that each 8-group is uniform, so row_mirror (lane^15)
// acts as xor8.  No LDS traffic, pure VALU.
// ---------------------------------------------------------------------------
static __device__ __forceinline__ float row16_sum(float x) {
#ifdef COSATTN_HAVE_DPP
    x += __builtin_bit_cast(float, __builtin_amdgcn_update_dpp(0, __builtin_bit_cast(int, x), 0xB1,  0xf, 0xf, true)); // quad_perm(1,0,3,2) : xor1
    x += __builtin_bit_cast(float, __builtin_amdgcn_update_dpp(0, __builtin_bit_cast(int, x), 0x4E,  0xf, 0xf, true)); // quad_perm(2,3,0,1) : xor2
    x += __builtin_bit_cast(float, __builtin_amdgcn_update_dpp(0, __builtin_bit_cast(int, x), 0x141, 0xf, 0xf, true)); // row_half_mirror    : xor4
    x += __builtin_bit_cast(float, __builtin_amdgcn_update_dpp(0, __builtin_bit_cast(int, x), 0x140, 0xf, 0xf, true)); // row_mirror         : xor8
#else
    x += __shfl_xor(x, 1, 32);
    x += __shfl_xor(x, 2, 32);
    x += __shfl_xor(x, 4, 32);
    x += __shfl_xor(x, 8, 32);
#endif
    return x;
}

// ===========================================================================
// Prep pass: one thread per (h,s) row.
//   kn[h][s][d] = f16( K / ||K|| )                (row-major, WMMA-B friendly)
//   vt[h][d][s] = f16( V )                        (transposed: PV B-frag = contiguous)
// ===========================================================================
__global__ __launch_bounds__(256) void cosattn_prep_kernel(
    const float* __restrict__ k,
    const float* __restrict__ v,
    _Float16* __restrict__ kn,
    _Float16* __restrict__ vt)
{
    const int row = blockIdx.x * 256 + threadIdx.x;   // 0 .. H*S-1
    const int h = row >> 12;                          // S_ = 4096
    const int s = row & (S_ - 1);

    // ---- normalize K row -> kn ----
    const float* kr = k + (size_t)row * D_;
    float kv[D_];
#pragma unroll
    for (int j = 0; j < D_ / 4; ++j) {
        float4 f = *(const float4*)(kr + j * 4);
        kv[j * 4 + 0] = f.x; kv[j * 4 + 1] = f.y; kv[j * 4 + 2] = f.z; kv[j * 4 + 3] = f.w;
    }
    float ss = 0.f;
#pragma unroll
    for (int i = 0; i < D_; ++i) ss += kv[i] * kv[i];
    const float inv = 1.0f / fmaxf(sqrtf(ss), 1e-12f);

    _Float16* ko = kn + (size_t)row * D_;
#pragma unroll
    for (int c = 0; c < 4; ++c) {
        v16h t;
#pragma unroll
        for (int i = 0; i < 16; ++i) t[i] = (_Float16)(kv[c * 16 + i] * inv);
        *(v16h*)(ko + c * 16) = t;
    }

    // ---- transpose V row -> vt (consecutive s => coalesced b16 stores per d) ----
    const float* vr = v + (size_t)row * D_;
#pragma unroll
    for (int d = 0; d < D_; ++d)
        vt[((size_t)(h * D_ + d)) * S_ + s] = (_Float16)vr[d];
}

// ===========================================================================
// Main flash-attention pass (fast path): f16 K/V prepared in workspace.
// Block = 128 threads = 4 independent waves; each wave owns TWO 16-row query
// sub-tiles (M=32) so the K/V fragments are reused twice and the two WMMA
// dependency chains fill each other's post-WMMA hazard slots.
// Fixed softmax shift c = scale (scores are bounded by scale) — no online max;
// l accumulated per-lane, reduced once at the end.  log2(e) folded into A so
// probabilities come from a single v_sub + raw v_exp_f32.
// ===========================================================================
__global__ __launch_bounds__(128) void cosattn_fa2_kernel(
    const float* __restrict__ q,
    const _Float16* __restrict__ kn,
    const _Float16* __restrict__ vt,
    const float* __restrict__ logit_scale,
    float* __restrict__ out)
{
    __shared__ _Float16 ldsP[4 * QT_ * 16 * PSTRH];   // per-wave, per-sub-tile P staging

    const int tid    = threadIdx.x;
    const int wave   = tid >> 5;
    const int lane   = tid & 31;
    const int lane16 = lane & 15;
    const int hi     = lane >> 4;

    const int h     = blockIdx.y;
    const int qbase = (blockIdx.x * 4 + wave) * (16 * QT_);

    const float scale = __expf(fminf(logit_scale[h], MAX_LOG_SCALE));
    const float c2    = scale * LOG2E;                // exp(e-scale) = exp2(e*log2e - c2)

    // ---- Load + normalize Q into A-fragments; fold scale*log2e into A ----
    // A frags (f16 16x32, our k order): lane m holds K 0-15 of row m, lane m+16 holds
    // K 16-31.  D=64 -> fragments a0 (d 0..31), a1 (d 32..63).
    const int c0 = hi * 16;
    v16h a0[QT_], a1[QT_];
#pragma unroll
    for (int qt = 0; qt < QT_; ++qt) {
        const float* qrow = q + ((size_t)h * S_ + qbase + qt * 16 + lane16) * D_;
        float qv[32];
#pragma unroll
        for (int c = 0; c < 4; ++c) {
            float4 f0 = *(const float4*)(qrow + c * 8 + (c >> 1) * 24 + c0);
            float4 f1 = *(const float4*)(qrow + c * 8 + (c >> 1) * 24 + c0 + 4);
            qv[c * 8 + 0] = f0.x; qv[c * 8 + 1] = f0.y; qv[c * 8 + 2] = f0.z; qv[c * 8 + 3] = f0.w;
            qv[c * 8 + 4] = f1.x; qv[c * 8 + 5] = f1.y; qv[c * 8 + 6] = f1.z; qv[c * 8 + 7] = f1.w;
        }
        float qss = 0.f;
#pragma unroll
        for (int i = 0; i < 32; ++i) qss += qv[i] * qv[i];
        qss += __shfl_xor(qss, 16, 32);               // lanes m / m+16 hold complementary halves
        const float qmul = c2 / fmaxf(sqrtf(qss), 1e-12f);   // |a| <= 11.6 : f16-safe
#pragma unroll
        for (int i = 0; i < 16; ++i) {
            a0[qt][i] = (_Float16)(qv[i]      * qmul);
            a1[qt][i] = (_Float16)(qv[16 + i] * qmul);
        }
    }

    float l_s[QT_][8];
    v8f acc[QT_][4];
    const v8f czero = {0,0,0,0,0,0,0,0};
#pragma unroll
    for (int qt = 0; qt < QT_; ++qt) {
#pragma unroll
        for (int r = 0; r < 8; ++r) l_s[qt][r] = 0.f;
#pragma unroll
        for (int t = 0; t < 4; ++t) acc[qt][t] = czero;
    }

    _Float16* lp = &ldsP[wave * (QT_ * 16 * PSTRH)];

    for (int kb = 0; kb < S_; kb += KB_) {
        // ---- K B-fragments straight from global f16 (contiguous 32B per lane) ----
        // B frags (32x16, our k order): lane n = key column n; lanes 0-15 hold d-base 0, lanes 16-31 d-base 16.
        const _Float16* kr0 = kn + ((size_t)h * S_ + kb + lane16) * D_ + hi * 16;
        const v16h b00 = *(const v16h*)(kr0);             // key tile0, d 0..31
        const v16h b10 = *(const v16h*)(kr0 + 32);        // key tile0, d 32..63
        const v16h b01 = *(const v16h*)(kr0 + 16 * D_);   // key tile1, d 0..31
        const v16h b11 = *(const v16h*)(kr0 + 16 * D_ + 32);

        // ---- scores + bounded softmax per query sub-tile ----
#pragma unroll
        for (int qt = 0; qt < QT_; ++qt) {
            v8f s0 = __builtin_amdgcn_wmma_f32_16x16x32_f16(false, a0[qt], false, b00, (short)0, czero, false, false);
            s0     = __builtin_amdgcn_wmma_f32_16x16x32_f16(false, a1[qt], false, b10, (short)0, s0,    false, false);
            v8f s1 = __builtin_amdgcn_wmma_f32_16x16x32_f16(false, a0[qt], false, b01, (short)0, czero, false, false);
            s1     = __builtin_amdgcn_wmma_f32_16x16x32_f16(false, a1[qt], false, b11, (short)0, s1,    false, false);

            _Float16* lpq = lp + qt * 16 * PSTRH;
#pragma unroll
            for (int r = 0; r < 8; ++r) {
                const float p0 = COSATTN_EXP2(s0[r] - c2);   // arg in [-2*c2, 0]: raw v_exp_f32 safe
                const float p1 = COSATTN_EXP2(s1[r] - c2);
                l_s[qt][r] += p0 + p1;                       // per-lane partial; reduced at end
                const int rr = r + 8 * hi;
                lpq[rr * PSTRH + lane16]      = (_Float16)p0;
                lpq[rr * PSTRH + 16 + lane16] = (_Float16)p1;
            }
        }

        __builtin_amdgcn_wave_barrier();                  // DS in-order within wave; fence compiler

        // ---- P: C-layout -> A-fragment via LDS (two aligned 16B chunks each) ----
        v16h pf[QT_];
#pragma unroll
        for (int qt = 0; qt < QT_; ++qt) {
            const _Float16* prow = lp + qt * 16 * PSTRH + lane16 * PSTRH;
            const v8h plo = *(const v8h*)(prow + hi * 16);       // K koff..koff+7
            const v8h phi = *(const v8h*)(prow + hi * 16 + 8);   // K koff+8..+15
            pf[qt] = __builtin_shufflevector(plo, phi,
                0,1,2,3,4,5,6,7, 8,9,10,11,12,13,14,15);
        }

        // ---- V B-fragments from transposed f16 vt (contiguous 32B), reused by both sub-tiles ----
#pragma unroll
        for (int t = 0; t < 4; ++t) {
            const int dim = t * 16 + lane16;
            const v16h vf = *(const v16h*)(vt + ((size_t)(h * D_ + dim)) * S_ + kb + hi * 16);
#pragma unroll
            for (int qt = 0; qt < QT_; ++qt)
                acc[qt][t] = __builtin_amdgcn_wmma_f32_16x16x32_f16(false, pf[qt], false, vf, (short)0, acc[qt][t], false, false);
        }

        __builtin_amdgcn_wave_barrier();                  // keep next iter's P stores behind reads
    }

    // ---- epilogue: reduce l across the 16 key-lanes once, divide, store ----
#pragma unroll
    for (int qt = 0; qt < QT_; ++qt) {
#pragma unroll
        for (int r = 0; r < 8; ++r) {
            const float lt = row16_sum(l_s[qt][r]);
            const float inv_l = 1.0f / lt;
            const int rr = r + 8 * hi;
            float* orow = out + ((size_t)h * S_ + qbase + qt * 16 + rr) * D_ + lane16;
            orow[0]  = acc[qt][0][r] * inv_l;
            orow[16] = acc[qt][1][r] * inv_l;
            orow[32] = acc[qt][2][r] * inv_l;
            orow[48] = acc[qt][3][r] * inv_l;
        }
    }
}

// ===========================================================================
// Fallback (no-workspace) kernel: self-contained version.
// ===========================================================================
#define VSTR 66
#define PSTR 34

__global__ __launch_bounds__(128) void cosattn_fa_kernel_fb(
    const float* __restrict__ q,
    const float* __restrict__ k,
    const float* __restrict__ v,
    const float* __restrict__ logit_scale,
    float* __restrict__ out)
{
    __shared__ float ldsV[KB_ * VSTR];
    __shared__ float ldsP[4 * 16 * PSTR];

    const int tid    = threadIdx.x;
    const int wave   = tid >> 5;
    const int lane   = tid & 31;
    const int lane16 = lane & 15;
    const int hi     = lane >> 4;

    const int h     = blockIdx.y;
    const int qbase = (blockIdx.x * 4 + wave) * 16;

    const float scale = __expf(fminf(logit_scale[h], MAX_LOG_SCALE));
    const float c2    = scale * LOG2E;

    const float* qrow = q + ((size_t)h * S_ + qbase + lane16) * D_;
    const int c0 = hi * 16;

    float qv[32];
#pragma unroll
    for (int c = 0; c < 4; ++c) {
        float4 f0 = *(const float4*)(qrow + c * 8 + (c >> 1) * 24 + c0);
        float4 f1 = *(const float4*)(qrow + c * 8 + (c >> 1) * 24 + c0 + 4);
        qv[c * 8 + 0] = f0.x; qv[c * 8 + 1] = f0.y; qv[c * 8 + 2] = f0.z; qv[c * 8 + 3] = f0.w;
        qv[c * 8 + 4] = f1.x; qv[c * 8 + 5] = f1.y; qv[c * 8 + 6] = f1.z; qv[c * 8 + 7] = f1.w;
    }
    float qss = 0.f;
#pragma unroll
    for (int i = 0; i < 32; ++i) qss += qv[i] * qv[i];
    qss += __shfl_xor(qss, 16, 32);
    const float qmul = c2 / fmaxf(sqrtf(qss), 1e-12f);

    v16h a0, a1;
#pragma unroll
    for (int i = 0; i < 16; ++i) {
        a0[i] = (_Float16)(qv[i]      * qmul);
        a1[i] = (_Float16)(qv[16 + i] * qmul);
    }

    float l_s[8];
#pragma unroll
    for (int r = 0; r < 8; ++r) l_s[r] = 0.f;
    v8f acc0 = {0,0,0,0,0,0,0,0};
    v8f acc1 = {0,0,0,0,0,0,0,0};
    v8f acc2 = {0,0,0,0,0,0,0,0};
    v8f acc3 = {0,0,0,0,0,0,0,0};
    const v8f czero = {0,0,0,0,0,0,0,0};

    float* lp = &ldsP[wave * 16 * PSTR];

    for (int kb = 0; kb < S_; kb += KB_) {
        {
            const float* vb = v + ((size_t)h * S_ + kb) * D_;
#pragma unroll
            for (int it = 0; it < 8; ++it) {
                int e   = tid + 128 * it;
                int row = e >> 5;
                int c2i = (e & 31) * 2;
                float2 val = *(const float2*)(vb + row * D_ + c2i);
                *(float2*)(&ldsV[row * VSTR + c2i]) = val;
            }
        }

        v16h b0[2], b1[2];
#pragma unroll
        for (int t = 0; t < 2; ++t) {
            const float* krow = k + ((size_t)h * S_ + kb + t * 16 + lane16) * D_ + hi * 16;
            float kv[32];
#pragma unroll
            for (int j = 0; j < 4; ++j) {
                float4 f0 = *(const float4*)(krow + j * 4);
                float4 f1 = *(const float4*)(krow + 32 + j * 4);
                kv[j * 4 + 0] = f0.x; kv[j * 4 + 1] = f0.y; kv[j * 4 + 2] = f0.z; kv[j * 4 + 3] = f0.w;
                kv[16 + j * 4 + 0] = f1.x; kv[16 + j * 4 + 1] = f1.y;
                kv[16 + j * 4 + 2] = f1.z; kv[16 + j * 4 + 3] = f1.w;
            }
            float kss = 0.f;
#pragma unroll
            for (int i = 0; i < 32; ++i) kss += kv[i] * kv[i];
            kss += __shfl_xor(kss, 16, 32);
            const float kinv = 1.0f / fmaxf(sqrtf(kss), 1e-12f);
#pragma unroll
            for (int i = 0; i < 16; ++i) {
                b0[t][i] = (_Float16)(kv[i]      * kinv);
                b1[t][i] = (_Float16)(kv[16 + i] * kinv);
            }
        }

        v8f s0 = __builtin_amdgcn_wmma_f32_16x16x32_f16(false, a0, false, b0[0], (short)0, czero, false, false);
        s0     = __builtin_amdgcn_wmma_f32_16x16x32_f16(false, a1, false, b1[0], (short)0, s0,    false, false);
        v8f s1 = __builtin_amdgcn_wmma_f32_16x16x32_f16(false, a0, false, b0[1], (short)0, czero, false, false);
        s1     = __builtin_amdgcn_wmma_f32_16x16x32_f16(false, a1, false, b1[1], (short)0, s1,    false, false);

#pragma unroll
        for (int r = 0; r < 8; ++r) {
            const float p0 = COSATTN_EXP2(s0[r] - c2);
            const float p1 = COSATTN_EXP2(s1[r] - c2);
            l_s[r] += p0 + p1;
            const int rr = r + 8 * hi;
            lp[rr * PSTR + lane16]      = p0;
            lp[rr * PSTR + 16 + lane16] = p1;
        }

        __syncthreads();

        v16h pf;
        {
            const float* prow = &lp[lane16 * PSTR];
            const int koff = hi * 16;
#pragma unroll
            for (int i = 0; i < 8; ++i) {
                pf[i]     = (_Float16)prow[koff + i];
                pf[8 + i] = (_Float16)prow[koff + 8 + i];
            }
        }

        {
            const int kbse = hi * 16;
#pragma unroll
            for (int t = 0; t < 4; ++t) {
                const int dim = t * 16 + lane16;
                v16h vf;
#pragma unroll
                for (int i = 0; i < 16; ++i)
                    vf[i] = (_Float16)ldsV[(kbse + i) * VSTR + dim];
                if      (t == 0) acc0 = __builtin_amdgcn_wmma_f32_16x16x32_f16(false, pf, false, vf, (short)0, acc0, false, false);
                else if (t == 1) acc1 = __builtin_amdgcn_wmma_f32_16x16x32_f16(false, pf, false, vf, (short)0, acc1, false, false);
                else if (t == 2) acc2 = __builtin_amdgcn_wmma_f32_16x16x32_f16(false, pf, false, vf, (short)0, acc2, false, false);
                else             acc3 = __builtin_amdgcn_wmma_f32_16x16x32_f16(false, pf, false, vf, (short)0, acc3, false, false);
            }
        }

        __syncthreads();
    }

#pragma unroll
    for (int r = 0; r < 8; ++r) {
        float lt = row16_sum(l_s[r]);
        const float inv_l = 1.0f / lt;
        const int rr = r + 8 * hi;
        float* orow = out + ((size_t)h * S_ + qbase + rr) * D_ + lane16;
        orow[0]  = acc0[r] * inv_l;
        orow[16] = acc1[r] * inv_l;
        orow[32] = acc2[r] * inv_l;
        orow[48] = acc3[r] * inv_l;
    }
}

// ===========================================================================
extern "C" void kernel_launch(void* const* d_in, const int* in_sizes, int n_in,
                              void* d_out, int out_size, void* d_ws, size_t ws_size,
                              hipStream_t stream) {
    const float* q  = (const float*)d_in[0];
    const float* k  = (const float*)d_in[1];
    const float* v  = (const float*)d_in[2];
    const float* ls = (const float*)d_in[3];
    float* out = (float*)d_out;
    (void)in_sizes; (void)n_in; (void)out_size;

    const size_t elems = (size_t)H_ * S_ * D_;         // per tensor
    const size_t need  = elems * sizeof(_Float16) * 2; // kn + vt = 16 MB

    if (ws_size >= need && d_ws != nullptr) {
        _Float16* kn = (_Float16*)d_ws;
        _Float16* vt = kn + elems;
        cosattn_prep_kernel<<<(H_ * S_) / 256, 256, 0, stream>>>(k, v, kn, vt);
        dim3 grid(S_ / (16 * QT_ * 4), H_);            // 32 x 16
        cosattn_fa2_kernel<<<grid, dim3(128), 0, stream>>>(q, kn, vt, ls, out);
    } else {
        dim3 grid(S_ / 64, H_);                        // 64 x 16
        cosattn_fa_kernel_fb<<<grid, dim3(128), 0, stream>>>(q, k, v, ls, out);
    }
}
